// VishwamaiAttention_82446192214074
// MI455X (gfx1250) — matrix operand order repaired and checked by reading, not compiled, not run on hardware
//
#include <hip/hip_runtime.h>

// Problem constants (match reference)
#define B_    2
#define S_    1024
#define H_    32
#define HK_   8
#define D_    128
#define HID_  (H_ * D_)   // 4096
#define KVD_  (HK_ * D_)  // 1024
#define REP_  (H_ / HK_)  // 4
#define THETA_ 10000.0f
#define SCALE_ 0.08838834764831845f  // 1/sqrt(128)

typedef __attribute__((ext_vector_type(16))) __bf16 v16bf;
typedef __attribute__((ext_vector_type(8)))  float  v8f;

struct __align__(16) U4 { unsigned int x, y, z, w; };

union Frag16 { v16bf v; U4 q[2]; unsigned int u[8]; unsigned short h[16]; };
union FragC  { v8f  v; float f[8]; };

__device__ __forceinline__ unsigned short f2bf(float f) {
  unsigned int u = __float_as_uint(f);
  u += 0x7FFFu + ((u >> 16) & 1u);   // round-to-nearest-even
  return (unsigned short)(u >> 16);
}

// gfx1250 async global->LDS copy: each lane moves 16 bytes.
// lds_off: per-lane LDS byte offset; g_off: per-lane byte offset vs base.
__device__ __forceinline__ void async_copy_b128(unsigned int lds_off,
                                                int g_off, const void* base) {
  asm volatile("global_load_async_to_lds_b128 %0, %1, %2"
               :: "v"(lds_off), "v"(g_off), "s"(base)
               : "memory");
}
__device__ __forceinline__ void wait_async0() {
  asm volatile("s_wait_asynccnt 0x0" ::: "memory");
}
__device__ __forceinline__ void wait_ds0() {
  asm volatile("s_wait_dscnt 0x0" ::: "memory");
}

// ---------------------------------------------------------------------------
// f32 -> bf16 elementwise conversion
// ---------------------------------------------------------------------------
__global__ void cvt_bf16_kernel(const float* __restrict__ src,
                                unsigned short* __restrict__ dst, int n) {
  int i = blockIdx.x * blockDim.x + threadIdx.x;
  if (i < n) dst[i] = f2bf(src[i]);
}

// ---------------------------------------------------------------------------
// RoPE + pack: src f32 [B, S, nh, D] -> dst bf16 [B, nh, S, D]
// ---------------------------------------------------------------------------
__global__ void rope_pack_kernel(const float* __restrict__ src,
                                 unsigned short* __restrict__ dst, int nh, int total) {
  int idx = blockIdx.x * blockDim.x + threadIdx.x;  // over B*S*nh*(D/2)
  if (idx >= total) return;
  int j = idx & (D_ / 2 - 1);
  int t = idx >> 6;
  int h = t % nh; t /= nh;
  int s = t % S_;
  int b = t / S_;
  float inv = __expf(-(2.0f * (float)j / (float)D_) * __logf(THETA_));
  float ang = (float)s * inv;
  float sn, cs;
  __sincosf(ang, &sn, &cs);
  const float* p = src + (((size_t)(b * S_ + s) * nh + h) * D_ + 2 * j);
  float xe = p[0], xo = p[1];
  unsigned short* q = dst + (((size_t)(b * nh + h) * S_ + s) * D_ + 2 * j);
  q[0] = f2bf(xe * cs - xo * sn);
  q[1] = f2bf(xe * sn + xo * cs);
}

// ---------------------------------------------------------------------------
// V transpose + pack: src f32 [B, S, HK, D] -> dst bf16 [B, HK, D, S]
// ---------------------------------------------------------------------------
__global__ void vtrans_kernel(const float* __restrict__ src,
                              unsigned short* __restrict__ dst, int total) {
  int idx = blockIdx.x * blockDim.x + threadIdx.x;  // over B*S*HK*D, d fastest
  if (idx >= total) return;
  int d = idx & (D_ - 1);
  int t = idx >> 7;
  int hk = t % HK_; t /= HK_;
  int s = t % S_;
  int b = t / S_;
  float v = src[((size_t)(b * S_ + s) * HK_ + hk) * D_ + d];
  dst[((size_t)(b * HK_ + hk) * D_ + d) * S_ + s] = f2bf(v);
}

// ---------------------------------------------------------------------------
// GEMM: C[M,N] f32 = A[M,K] bf16 @ B[K,N] bf16.  Block tile 128x128x32,
// 256 threads = 8 waves, each wave computes 32x64 via v_wmma_f32_16x16x32_bf16.
// A tile staged with async global->LDS DMA; B tile transposed via u16 scatter.
// ---------------------------------------------------------------------------
#define GBM 128
#define GBN 128
#define GBK 32
#define APAD 40  // 80B row stride: 16B-aligned, bank-conflict free (20 dwords)

__global__ __launch_bounds__(256)
void gemm_bf16_kernel(const unsigned short* __restrict__ A,
                      const unsigned short* __restrict__ B,
                      float* __restrict__ C, int M, int N, int K) {
  __shared__ alignas(16) unsigned short As[GBM][APAD];
  __shared__ alignas(16) unsigned short Bs[GBN][APAD];  // transposed: Bs[n][k]

  const int tid = threadIdx.x;
  const int wave = tid >> 5;
  const int lane = tid & 31;
  const int lanelo = lane & 15;
  const int hi = lane >> 4;
  const int wm = (wave & 3) * 32;   // wave M offset in block tile
  const int wn = (wave >> 2) * 64;  // wave N offset in block tile
  const int m0 = blockIdx.y * GBM;
  const int n0 = blockIdx.x * GBN;

  const unsigned int asBase = (unsigned int)(size_t)&As[0][0];

  FragC c[2][4];
#pragma unroll
  for (int mi = 0; mi < 2; ++mi)
#pragma unroll
    for (int ni = 0; ni < 4; ++ni)
#pragma unroll
      for (int g = 0; g < 8; ++g) c[mi][ni].f[g] = 0.0f;

  for (int k0 = 0; k0 < K; k0 += GBK) {
    // Stage A tile (128 rows x 64B) with async DMA: 512 lane-chunks of 16B.
#pragma unroll
    for (int p = 0; p < 2; ++p) {
      int idx = wave * 64 + p * 32 + lane;  // 0..511
      int r  = idx >> 2;                    // row 0..127
      int cc = idx & 3;                     // 16B chunk within row
      unsigned int lds_off = asBase + (unsigned int)(r * (APAD * 2) + cc * 16);
      int g_off = ((m0 + r) * K + k0) * 2 + cc * 16;
      async_copy_b128(lds_off, g_off, A);
    }
    // Stage B tile (32 x 128) transposed into Bs[n][k].
#pragma unroll
    for (int p = 0; p < 8; ++p) {
      int linear = p * 256 + tid;       // 0..2047
      int kk = linear >> 6;             // 0..31
      int c2 = linear & 63;
      unsigned int val =
          *(const unsigned int*)(B + (size_t)(k0 + kk) * N + n0 + c2 * 2);
      Bs[c2 * 2 + 0][kk] = (unsigned short)(val & 0xFFFFu);
      Bs[c2 * 2 + 1][kk] = (unsigned short)(val >> 16);
    }
    wait_async0();
    __syncthreads();

    Frag16 a[2];
#pragma unroll
    for (int mi = 0; mi < 2; ++mi) {
      const unsigned short* arow = &As[wm + mi * 16 + lanelo][0];
      a[mi].q[0] = *(const U4*)(arow + hi * 8);        // K = kb..kb+7
      a[mi].q[1] = *(const U4*)(arow + 16 + hi * 8);   // K = 16+kb..+7
    }
#pragma unroll
    for (int ni = 0; ni < 4; ++ni) {
      Frag16 b;
      const unsigned short* brow = &Bs[wn + ni * 16 + lanelo][0];
      b.q[0] = *(const U4*)(brow + hi * 8);            // K = hi*8..+7
      b.q[1] = *(const U4*)(brow + hi * 8 + 16);       // K = 16+hi*8..+7
#pragma unroll
      for (int mi = 0; mi < 2; ++mi)
        c[mi][ni].v = __builtin_amdgcn_wmma_f32_16x16x32_bf16(
            false, a[mi].v, false, b.v, (short)0, c[mi][ni].v, false, false);
    }
    __syncthreads();
  }

#pragma unroll
  for (int mi = 0; mi < 2; ++mi)
#pragma unroll
    for (int ni = 0; ni < 4; ++ni)
#pragma unroll
      for (int g = 0; g < 8; ++g) {
        int row = m0 + wm + mi * 16 + g + hi * 8;
        int col = n0 + wn + ni * 16 + lanelo;
        C[(size_t)row * N + col] = c[mi][ni].f[g];
      }
}

// ---------------------------------------------------------------------------
// Flash attention: qh bf16 [B,H,S,D], kh bf16 [B,HK,S,D], vT bf16 [B,HK,D,S]
// -> attnb bf16 [B,S,H*D].  Grid (B*H, S/64), block 128 (4 waves).
// Each wave owns 16 query rows; key blocks of 64 with online softmax.
// K/V tiles staged with async global->LDS DMA.
// ---------------------------------------------------------------------------
#define KPAD 136  // Ks row stride (elems): 272B, 16B mult, conflict-free
#define VPAD 72   // Vs/Ps row stride (elems): 144B, 16B mult, conflict-free

__global__ __launch_bounds__(128)
void flash_attn_kernel(const unsigned short* __restrict__ qh,
                       const unsigned short* __restrict__ kh,
                       const unsigned short* __restrict__ vT,
                       unsigned short* __restrict__ attnb) {
  __shared__ alignas(16) unsigned short Ks[64][KPAD];     // [key][d]
  __shared__ alignas(16) unsigned short Vs[D_][VPAD];     // [d][key] (from vT)
  __shared__ alignas(16) unsigned short Ps[4][16][VPAD];  // per-wave P [row][key]

  const int tid = threadIdx.x;
  const int wid = tid >> 5;
  const int lane = tid & 31;
  const int lanelo = lane & 15;
  const int hi = lane >> 4;

  const int bh = blockIdx.x;
  const int b = bh / H_;
  const int h = bh % H_;
  const int hk = h / REP_;
  const int qb = blockIdx.y;
  const int q0 = qb * 64 + wid * 16;

  const unsigned int ksBase = (unsigned int)(size_t)&Ks[0][0];
  const unsigned int vsBase = (unsigned int)(size_t)&Vs[0][0];

  // Q fragments (A-matrix layout), loaded once from global.
  const unsigned short* qrow =
      qh + ((size_t)(b * H_ + h) * S_ + q0 + lanelo) * D_;
  Frag16 aQ[4];
#pragma unroll
  for (int kd = 0; kd < 4; ++kd) {
    aQ[kd].q[0] = *(const U4*)(qrow + kd * 32 + hi * 8);
    aQ[kd].q[1] = *(const U4*)(qrow + kd * 32 + hi * 8 + 16);
  }

  FragC o[8];
#pragma unroll
  for (int nd = 0; nd < 8; ++nd)
#pragma unroll
    for (int g = 0; g < 8; ++g) o[nd].f[g] = 0.0f;
  float m_run[8], l_run[8];
#pragma unroll
  for (int g = 0; g < 8; ++g) { m_run[g] = -1e30f; l_run[g] = 0.0f; }

  const unsigned short* kbase_g = kh + (size_t)(b * HK_ + hk) * S_ * D_;
  const unsigned short* vbase_g = vT + (size_t)(b * HK_ + hk) * D_ * S_;

  for (int kb = 0; kb <= qb; ++kb) {
    __syncthreads();  // previous iteration's Ks/Vs reads complete
    // Stage K tile: 64 rows x 256B  -> 1024 lane-chunks of 16B (8 per thread)
#pragma unroll
    for (int p = 0; p < 8; ++p) {
      int idx = p * 128 + tid;          // 0..1023
      int r = idx >> 4;                 // key row 0..63
      int cc = idx & 15;                // 16B chunk (d)
      unsigned int lds_off = ksBase + (unsigned int)(r * (KPAD * 2) + cc * 16);
      int g_off = (kb * 64 + r) * (D_ * 2) + cc * 16;
      async_copy_b128(lds_off, g_off, kbase_g);
    }
    // Stage Vt tile: 128 rows (d) x 128B -> 1024 lane-chunks of 16B
#pragma unroll
    for (int p = 0; p < 8; ++p) {
      int idx = p * 128 + tid;          // 0..1023
      int d = idx >> 3;                 // d row 0..127
      int cc = idx & 7;                 // 16B chunk (key)
      unsigned int lds_off = vsBase + (unsigned int)(d * (VPAD * 2) + cc * 16);
      int g_off = (d * S_ + kb * 64) * 2 + cc * 16;
      async_copy_b128(lds_off, g_off, vbase_g);
    }
    wait_async0();
    __syncthreads();

    // S = Q @ K^T  (4 N-tiles of 16 keys, K-dim = D = 4 x 32)
    FragC s[4];
#pragma unroll
    for (int nt = 0; nt < 4; ++nt) {
#pragma unroll
      for (int g = 0; g < 8; ++g) s[nt].f[g] = 0.0f;
      const unsigned short* krow = &Ks[nt * 16 + lanelo][0];
#pragma unroll
      for (int kd = 0; kd < 4; ++kd) {
        Frag16 bk;
        bk.q[0] = *(const U4*)(krow + kd * 32 + hi * 8);
        bk.q[1] = *(const U4*)(krow + kd * 32 + hi * 8 + 16);
        s[nt].v = __builtin_amdgcn_wmma_f32_16x16x32_bf16(
            false, aQ[kd].v, false, bk.v, (short)0, s[nt].v, false, false);
      }
    }
    // scale + causal mask
#pragma unroll
    for (int nt = 0; nt < 4; ++nt)
#pragma unroll
      for (int g = 0; g < 8; ++g) {
        float val = s[nt].f[g] * SCALE_;
        if (kb == qb) {
          int grow = q0 + g + hi * 8;
          int gcol = kb * 64 + nt * 16 + lanelo;
          if (gcol > grow) val = -1e30f;
        }
        s[nt].f[g] = val;
      }
    // Online softmax per row (row g+8*hi; 16 cols across lanes of half-wave)
#pragma unroll
    for (int g = 0; g < 8; ++g) {
      float mx = s[0].f[g];
#pragma unroll
      for (int nt = 1; nt < 4; ++nt) mx = fmaxf(mx, s[nt].f[g]);
#pragma unroll
      for (int msk = 8; msk >= 1; msk >>= 1)
        mx = fmaxf(mx, __shfl_xor(mx, msk, 32));
      float mnew = fmaxf(m_run[g], mx);
      float alpha = __expf(m_run[g] - mnew);
      m_run[g] = mnew;
      float rsum = 0.0f;
#pragma unroll
      for (int nt = 0; nt < 4; ++nt) {
        float p = __expf(s[nt].f[g] - mnew);
        s[nt].f[g] = p;
        rsum += p;
      }
#pragma unroll
      for (int msk = 8; msk >= 1; msk >>= 1)
        rsum += __shfl_xor(rsum, msk, 32);
      l_run[g] = l_run[g] * alpha + rsum;
#pragma unroll
      for (int nd = 0; nd < 8; ++nd) o[nd].f[g] *= alpha;
    }
    // Write P tile (C-layout -> wave-private LDS) then re-read in A-layout.
    // Wave-local fence suffices: LDS ops are in-order per wave.
#pragma unroll
    for (int nt = 0; nt < 4; ++nt)
#pragma unroll
      for (int g = 0; g < 8; ++g)
        Ps[wid][g + hi * 8][nt * 16 + lanelo] = f2bf(s[nt].f[g]);
    wait_ds0();

    Frag16 pf[2];
    const unsigned short* prow = &Ps[wid][lanelo][0];
#pragma unroll
    for (int kt = 0; kt < 2; ++kt) {
      pf[kt].q[0] = *(const U4*)(prow + kt * 32 + hi * 8);
      pf[kt].q[1] = *(const U4*)(prow + kt * 32 + hi * 8 + 16);
    }
    // O += P @ V  (8 d-tiles of 16, K-dim = 64 keys = 2 x 32)
#pragma unroll
    for (int nd = 0; nd < 8; ++nd) {
      const unsigned short* vrow = &Vs[nd * 16 + lanelo][0];
#pragma unroll
      for (int kt = 0; kt < 2; ++kt) {
        Frag16 vf;
        vf.q[0] = *(const U4*)(vrow + kt * 32 + hi * 8);
        vf.q[1] = *(const U4*)(vrow + kt * 32 + hi * 8 + 16);
        o[nd].v = __builtin_amdgcn_wmma_f32_16x16x32_bf16(
            false, pf[kt].v, false, vf.v, (short)0, o[nd].v, false, false);
      }
    }
  }

  // Epilogue: normalize and store bf16 [b*S+row][h*D + col]
#pragma unroll
  for (int g = 0; g < 8; ++g) {
    float inv_l = 1.0f / l_run[g];
    int row = q0 + g + hi * 8;
#pragma unroll
    for (int nd = 0; nd < 8; ++nd) {
      attnb[(size_t)(b * S_ + row) * HID_ + h * D_ + nd * 16 + lanelo] =
          f2bf(o[nd].f[g] * inv_l);
    }
  }
}

// ---------------------------------------------------------------------------
// Host-side orchestration
// ---------------------------------------------------------------------------
extern "C" void kernel_launch(void* const* d_in, const int* in_sizes, int n_in,
                              void* d_out, int out_size, void* d_ws,
                              size_t ws_size, hipStream_t stream) {
  (void)in_sizes; (void)n_in; (void)out_size; (void)ws_size;
  const float* x  = (const float*)d_in[0];
  const float* wq = (const float*)d_in[1];
  const float* wk = (const float*)d_in[2];
  const float* wv = (const float*)d_in[3];
  const float* wo = (const float*)d_in[4];
  // d_in[5] = attention_mask (pure causal; applied analytically in-kernel)

  const int M = B_ * S_;  // 2048

  char* ws = (char*)d_ws;
  size_t off = 0;
  auto nxt = [&](size_t bytes) -> void* {
    void* p = ws + off;
    off += (bytes + 255) & ~(size_t)255;
    return p;
  };
  unsigned short* xb  = (unsigned short*)nxt((size_t)M * HID_ * 2);
  unsigned short* wqb = (unsigned short*)nxt((size_t)HID_ * HID_ * 2);
  unsigned short* wkb = (unsigned short*)nxt((size_t)HID_ * KVD_ * 2);
  unsigned short* wvb = (unsigned short*)nxt((size_t)HID_ * KVD_ * 2);
  unsigned short* wob = (unsigned short*)nxt((size_t)HID_ * HID_ * 2);
  float* qf = (float*)nxt((size_t)M * HID_ * 4);
  float* kf = (float*)nxt((size_t)M * KVD_ * 4);
  float* vf = (float*)nxt((size_t)M * KVD_ * 4);
  unsigned short* qhp = (unsigned short*)nxt((size_t)M * HID_ * 2);
  unsigned short* khp = (unsigned short*)nxt((size_t)M * KVD_ * 2);
  unsigned short* vtp = (unsigned short*)nxt((size_t)M * KVD_ * 2);
  unsigned short* ab  = (unsigned short*)nxt((size_t)M * HID_ * 2);

  // 1) f32 -> bf16 conversions
  {
    int n;
    n = M * HID_;      cvt_bf16_kernel<<<(n + 255) / 256, 256, 0, stream>>>(x,  xb,  n);
    n = HID_ * HID_;   cvt_bf16_kernel<<<(n + 255) / 256, 256, 0, stream>>>(wq, wqb, n);
    n = HID_ * KVD_;   cvt_bf16_kernel<<<(n + 255) / 256, 256, 0, stream>>>(wk, wkb, n);
    n = HID_ * KVD_;   cvt_bf16_kernel<<<(n + 255) / 256, 256, 0, stream>>>(wv, wvb, n);
    n = HID_ * HID_;   cvt_bf16_kernel<<<(n + 255) / 256, 256, 0, stream>>>(wo, wob, n);
  }

  // 2) QKV projections (WMMA GEMMs)
  gemm_bf16_kernel<<<dim3(HID_ / GBN, M / GBM), 256, 0, stream>>>(
      xb, wqb, qf, M, HID_, HID_);
  gemm_bf16_kernel<<<dim3(KVD_ / GBN, M / GBM), 256, 0, stream>>>(
      xb, wkb, kf, M, KVD_, HID_);
  gemm_bf16_kernel<<<dim3(KVD_ / GBN, M / GBM), 256, 0, stream>>>(
      xb, wvb, vf, M, KVD_, HID_);

  // 3) RoPE + head-major packing; V transpose for WMMA B-fragments
  {
    int n;
    n = B_ * S_ * H_ * (D_ / 2);
    rope_pack_kernel<<<(n + 255) / 256, 256, 0, stream>>>(qf, qhp, H_, n);
    n = B_ * S_ * HK_ * (D_ / 2);
    rope_pack_kernel<<<(n + 255) / 256, 256, 0, stream>>>(kf, khp, HK_, n);
    n = B_ * S_ * HK_ * D_;
    vtrans_kernel<<<(n + 255) / 256, 256, 0, stream>>>(vf, vtp, n);
  }

  // 4) Causal flash attention with WMMA (QK^T and PV on matrix cores)
  flash_attn_kernel<<<dim3(B_ * H_, S_ / 64), 128, 0, stream>>>(
      qhp, khp, vtp, ab);

  // 5) Output projection straight into d_out (f32)
  gemm_bf16_kernel<<<dim3(HID_ / GBN, M / GBM), 256, 0, stream>>>(
      ab, wob, (float*)d_out, M, HID_, HID_);
}
